// TransformerBlock_18202071400825
// MI455X (gfx1250) — matrix operand order repaired and checked by reading, not compiled, not run on hardware
//
#include <hip/hip_runtime.h>

// ---------------------------------------------------------------------------
// Types and WMMA helpers (CDNA5 / gfx1250, wave32)
// ---------------------------------------------------------------------------
typedef __attribute__((ext_vector_type(16))) __bf16 v16bf;
typedef __attribute__((ext_vector_type(8)))  float  v8f;
typedef __bf16 bf16;

union BF16x16 { uint4 u[2]; v16bf v; };

__device__ __forceinline__ v8f wmma_bf16(v16bf a, v16bf b, v8f c) {
  // D = A(16x32 bf16) x B(32x16 bf16) + C(16x16 f32)
  return __builtin_amdgcn_wmma_f32_16x16x32_bf16(
      /*neg_a=*/false, a, /*neg_b=*/false, b,
      /*c_mod=*/(short)0, c, /*reuse_a=*/false, /*reuse_b=*/false);
}

__device__ __forceinline__ float gelu_tanh(float x) {
  const float c = 0.7978845608028654f;
  float t = tanhf(c * (x + 0.044715f * x * x * x));
  return 0.5f * x * (1.0f + t);
}

// ---------------------------------------------------------------------------
// Tensor Data Mover (gfx1250). 2-D tile of bf16: rows x cols elements,
// source row stride in elements, destination LDS contiguous.
// Descriptor bit layout per CDNA5 ISA (D# groups 0/1).
// This toolchain exposes the 6-argument builtin:
//   (uint32x4 g0, int32x8 g1, int32x4 g2, int32x4 g3, int32x8 g4, i32 cpol)
// ---------------------------------------------------------------------------
#if __has_builtin(__builtin_amdgcn_tensor_load_to_lds)
#define HAVE_TDM 1
#else
#define HAVE_TDM 0
#endif

#if HAVE_TDM
typedef __attribute__((ext_vector_type(4))) unsigned int u32x4_t;
typedef __attribute__((ext_vector_type(8))) int i32x8_t;
typedef __attribute__((ext_vector_type(4))) int i32x4_t;

__device__ __forceinline__ void tdm_load_2d_bf16(unsigned lds_addr, const void* gptr,
                                                 unsigned rows, unsigned cols,
                                                 unsigned row_stride_elems) {
  unsigned long long ga = (unsigned long long)gptr;
  u32x4_t g0;
  g0[0] = 1u;                                           // count=1, user mode
  g0[1] = lds_addr;                                     // LDS byte address
  g0[2] = (unsigned)(ga & 0xffffffffu);                 // global_addr[31:0]
  g0[3] = (unsigned)((ga >> 32) & 0x01ffffffu) | (2u << 30);  // [56:32] | type=2
  i32x8_t g1;
  g1[0] = (int)(1u << 16);                              // data_size=1 -> 2 bytes
  g1[1] = (int)((cols & 0xffffu) << 16);                // tensor_dim0[15:0]
  g1[2] = (int)(((cols >> 16) & 0xffffu) | ((rows & 0xffffu) << 16)); // dim0 hi | dim1 lo
  g1[3] = (int)(((rows >> 16) & 0xffffu) | ((cols & 0xffffu) << 16)); // dim1 hi | tile_dim0
  g1[4] = (int)(rows & 0xffffu);                        // tile_dim1 | tile_dim2=0
  g1[5] = (int)row_stride_elems;                        // tensor_dim0_stride[31:0]
  g1[6] = 0;                                            // stride hi | dim1_stride lo
  g1[7] = 0;
  i32x4_t z4 = {0, 0, 0, 0};
  i32x8_t z8 = {0, 0, 0, 0, 0, 0, 0, 0};
  __builtin_amdgcn_tensor_load_to_lds(g0, g1, z4, z4, z8, 0);
}
#endif

#if __has_builtin(__builtin_amdgcn_s_wait_tensorcnt)
#define TDM_WAIT(n) __builtin_amdgcn_s_wait_tensorcnt(n)
#else
#define TDM_WAIT(n)
#endif

// Problem constants
#define BB 2
#define SS 2048
#define DD 1024
#define HH 16
#define HD 64
#define EE 8
#define FF 4096
#define TT 4096  // B*S

// ---------------------------------------------------------------------------
// Utility kernels
// ---------------------------------------------------------------------------
__global__ void init_counts_kernel(int* counts) {
  if (threadIdx.x < EE) counts[threadIdx.x] = 0;
}

// flat f32 -> bf16
__global__ void cvt_bf16_kernel(const float* __restrict__ in, bf16* __restrict__ out, long n) {
  long i = (long)blockIdx.x * blockDim.x + threadIdx.x;
  long stride = (long)gridDim.x * blockDim.x;
  for (; i < n; i += stride) out[i] = (bf16)in[i];
}

// batched transpose+convert: in [batch][rows][cols] f32 -> out [batch][cols][rows] bf16
__global__ void cvt_t_bf16_kernel(const float* __restrict__ in, bf16* __restrict__ out,
                                  int rows, int cols) {
  long base = (long)blockIdx.y * rows * cols;
  long n = (long)rows * cols;
  long i = (long)blockIdx.x * blockDim.x + threadIdx.x;
  long stride = (long)gridDim.x * blockDim.x;
  for (; i < n; i += stride) {
    long r = i / cols, c = i % cols;
    out[base + c * (long)rows + r] = (bf16)in[base + i];
  }
}

// LayerNorm over D=1024; one block (128 thr) per row. outf optional.
__global__ void ln_kernel(const float* __restrict__ x, const float* __restrict__ g,
                          const float* __restrict__ be, bf16* __restrict__ outb,
                          float* __restrict__ outf) {
  int row = blockIdx.x;
  int tid = threadIdx.x;  // 128
  const float* xr = x + (long)row * DD;
  float s = 0.f, ss = 0.f;
  for (int i = tid; i < DD; i += 128) { float v = xr[i]; s += v; ss += v * v; }
  for (int off = 16; off > 0; off >>= 1) { s += __shfl_xor(s, off); ss += __shfl_xor(ss, off); }
  __shared__ float red[8];
  int wave = tid >> 5;
  if ((tid & 31) == 0) { red[wave] = s; red[4 + wave] = ss; }
  __syncthreads();
  s  = red[0] + red[1] + red[2] + red[3];
  ss = red[4] + red[5] + red[6] + red[7];
  float mean = s * (1.f / DD);
  float var  = ss * (1.f / DD) - mean * mean;
  float rstd = rsqrtf(var + 1e-5f);
  for (int i = tid; i < DD; i += 128) {
    float v = (xr[i] - mean) * rstd * g[i] + be[i];
    outb[(long)row * DD + i] = (bf16)v;
    if (outf) outf[(long)row * DD + i] = v;
  }
}

// ---------------------------------------------------------------------------
// GEMM: C[M][N] = A[M][K](bf16) * W^T + bias (+ residual), W stored [N][K] bf16
// Double-buffered A/B register sets so loads overlap WMMA issue.
// ---------------------------------------------------------------------------
__global__ void __launch_bounds__(128) gemm_abt_kernel(
    const bf16* __restrict__ A, const bf16* __restrict__ Wt,
    const float* __restrict__ bias, const float* __restrict__ residual,
    float* __restrict__ C, int Kdim, int Ndim) {
  const int lane = threadIdx.x & 31;
  const int wave = threadIdx.x >> 5;
  const int half = lane >> 4;
  const int nlo  = lane & 15;
  const int kset = half * 8;
  const int m0 = blockIdx.x * 16;
  const int n0 = blockIdx.y * 256 + wave * 64;
  const bf16* Arow = A + (long)(m0 + nlo) * Kdim;
  const v8f vz = {0.f,0.f,0.f,0.f,0.f,0.f,0.f,0.f};
  v8f acc[4] = {vz, vz, vz, vz};

  BF16x16 a0, a1, b0[4], b1[4];
  auto loadA = [&](BF16x16& a, int k0) {
    a.u[0] = *(const uint4*)(Arow + k0 + 2 * kset);
    a.u[1] = *(const uint4*)(Arow + k0 + 8 + 2 * kset);
  };
  auto loadB = [&](BF16x16* b, int k0) {
#pragma unroll
    for (int s2 = 0; s2 < 4; ++s2) {
      const bf16* br = Wt + (long)(n0 + s2 * 16 + nlo) * Kdim + k0 + half * 16;
      b[s2].u[0] = ((const uint4*)br)[0];
      b[s2].u[1] = ((const uint4*)br)[1];
    }
  };

  loadA(a0, 0);  loadB(b0, 0);
  loadA(a1, 32); loadB(b1, 32);
  for (int k0 = 64; k0 < Kdim; k0 += 64) {
#pragma unroll
    for (int s2 = 0; s2 < 4; ++s2) acc[s2] = wmma_bf16(a0.v, b0[s2].v, acc[s2]);
    loadA(a0, k0); loadB(b0, k0);
#pragma unroll
    for (int s2 = 0; s2 < 4; ++s2) acc[s2] = wmma_bf16(a1.v, b1[s2].v, acc[s2]);
    loadA(a1, k0 + 32); loadB(b1, k0 + 32);
  }
#pragma unroll
  for (int s2 = 0; s2 < 4; ++s2) acc[s2] = wmma_bf16(a0.v, b0[s2].v, acc[s2]);
#pragma unroll
  for (int s2 = 0; s2 < 4; ++s2) acc[s2] = wmma_bf16(a1.v, b1[s2].v, acc[s2]);

#pragma unroll
  for (int s2 = 0; s2 < 4; ++s2) {
    int col = n0 + s2 * 16 + nlo;
    float bv = bias[col];
#pragma unroll
    for (int r = 0; r < 8; ++r) {
      int row = m0 + r + half * 8;
      long off = (long)row * Ndim + col;
      float v = acc[s2][r] + bv;
      if (residual) v += residual[off];
      C[off] = v;
    }
  }
}

// ---------------------------------------------------------------------------
// Split qkv[T][3D] f32 -> qh/kh [B*H][S][64] bf16, vT [B*H][64][S] bf16.
// ---------------------------------------------------------------------------
__global__ void split_qkv_kernel(const float* __restrict__ qkv, bf16* __restrict__ qh,
                                 bf16* __restrict__ kh, bf16* __restrict__ vT) {
  long n = (long)TT * DD;
  long stride = (long)gridDim.x * blockDim.x;
  for (long i = (long)blockIdx.x * blockDim.x + threadIdx.x; i < n; i += stride) {
    int t = (int)(i >> 10);
    int dcol = (int)(i & (DD - 1));
    int h = dcol >> 6, dh = dcol & 63;
    int b = t >> 11, sI = t & (SS - 1);
    long qoff = (long)t * (3 * DD) + dcol;
    long hoff = (((long)(b * HH + h)) * SS + sI) * HD + dh;
    qh[hoff] = (bf16)(qkv[qoff] * 0.125f);
    kh[hoff] = (bf16)(qkv[qoff + DD]);
    vT[(((long)(b * HH + h)) * HD + dh) * SS + sI] = (bf16)(qkv[qoff + 2 * DD]);
  }
}

// ---------------------------------------------------------------------------
// Flash attention with TDM-streamed K/V tiles (double-buffered on TENSORcnt).
// ---------------------------------------------------------------------------
__global__ void __launch_bounds__(32) attn_kernel(const bf16* __restrict__ qh,
                                                  const bf16* __restrict__ kh,
                                                  const bf16* __restrict__ vT,
                                                  bf16* __restrict__ ctxb) {
  const int bh = blockIdx.x;
  const int q0 = blockIdx.y * 16;
  const int lane = threadIdx.x;
  const int half = lane >> 4;
  const int nlo  = lane & 15;
  const int kset = half * 8;
  __shared__ bf16 pls[16 * 32];
#if HAVE_TDM
  __shared__ bf16 kbuf[2][32 * 64];   // 32 keys x 64 hd
  __shared__ bf16 vbuf[2][64 * 32];   // 64 hd x 32 keys
#endif

  const bf16* qrow = qh + ((long)bh * SS + q0 + nlo) * HD;
  BF16x16 a0, a1;
  a0.u[0] = *(const uint4*)(qrow + 2 * kset);
  a0.u[1] = *(const uint4*)(qrow + 8 + 2 * kset);
  a1.u[0] = *(const uint4*)(qrow + 32 + 2 * kset);
  a1.u[1] = *(const uint4*)(qrow + 40 + 2 * kset);

  const v8f vz = {0.f,0.f,0.f,0.f,0.f,0.f,0.f,0.f};
  v8f acc[4] = {vz, vz, vz, vz};
  float mrun[8], lrun[8];
#pragma unroll
  for (int r = 0; r < 8; ++r) { mrun[r] = -1e30f; lrun[r] = 0.f; }

  const int ktend = q0 + 16;
#if HAVE_TDM
  int cur = 0;
  tdm_load_2d_bf16((unsigned)(unsigned long long)&kbuf[0][0],
                   kh + ((long)bh * SS) * HD, 32, 64, 64);
  tdm_load_2d_bf16((unsigned)(unsigned long long)&vbuf[0][0],
                   vT + (long)bh * HD * SS, 64, 32, SS);
#endif
  for (int kt = 0; kt < ktend; kt += 32) {
#if HAVE_TDM
    if (kt + 32 < ktend) {
      tdm_load_2d_bf16((unsigned)(unsigned long long)&kbuf[cur ^ 1][0],
                       kh + ((long)bh * SS + kt + 32) * HD, 32, 64, 64);
      tdm_load_2d_bf16((unsigned)(unsigned long long)&vbuf[cur ^ 1][0],
                       vT + (long)bh * HD * SS + kt + 32, 64, 32, SS);
      TDM_WAIT(2);
    } else {
      TDM_WAIT(0);
    }
    const bf16* ksrc = &kbuf[cur][0];
    const bf16* vsrc = &vbuf[cur][0];
    const int vstride = 32;
#else
    const bf16* ksrc = kh + ((long)bh * SS + kt) * HD;
    const bf16* vsrc = vT + (long)bh * HD * SS + kt;
    const int vstride = SS;
#endif
    v8f sv[2];
#pragma unroll
    for (int j = 0; j < 2; ++j) {
      const bf16* kr = ksrc + (j * 16 + nlo) * 64 + half * 16;
      BF16x16 b0, b1;
      b0.u[0] = ((const uint4*)kr)[0];        b0.u[1] = ((const uint4*)kr)[1];
      b1.u[0] = ((const uint4*)(kr + 32))[0]; b1.u[1] = ((const uint4*)(kr + 32))[1];
      v8f sj = vz;
      sj = wmma_bf16(a0.v, b0.v, sj);
      sj = wmma_bf16(a1.v, b1.v, sj);
      sv[j] = sj;
    }
#pragma unroll
    for (int r = 0; r < 8; ++r) {
      int mrow = q0 + r + half * 8;
      float s0 = sv[0][r]; if (kt + nlo > mrow)      s0 = -1e9f;
      float s1 = sv[1][r]; if (kt + 16 + nlo > mrow) s1 = -1e9f;
      float mx = fmaxf(s0, s1);
      mx = fmaxf(mx, __shfl_xor(mx, 1));
      mx = fmaxf(mx, __shfl_xor(mx, 2));
      mx = fmaxf(mx, __shfl_xor(mx, 4));
      mx = fmaxf(mx, __shfl_xor(mx, 8));
      float nm = fmaxf(mrun[r], mx);
      float sc = __expf(mrun[r] - nm);
      mrun[r] = nm;
      float p0 = __expf(s0 - nm);
      float p1 = __expf(s1 - nm);
      float rs = p0 + p1;
      rs += __shfl_xor(rs, 1); rs += __shfl_xor(rs, 2);
      rs += __shfl_xor(rs, 4); rs += __shfl_xor(rs, 8);
      lrun[r] = lrun[r] * sc + rs;
#pragma unroll
      for (int s2 = 0; s2 < 4; ++s2) acc[s2][r] *= sc;
      int rowl = r + half * 8;
      pls[rowl * 32 + nlo]      = (bf16)p0;
      pls[rowl * 32 + 16 + nlo] = (bf16)p1;
    }
    __syncthreads();
    BF16x16 ap;
    const bf16* pr = pls + nlo * 32;
    ap.u[0] = *(const uint4*)(pr + 2 * kset);
    ap.u[1] = *(const uint4*)(pr + 8 + 2 * kset);
#pragma unroll
    for (int s2 = 0; s2 < 4; ++s2) {
      const bf16* vr = vsrc + (s2 * 16 + nlo) * vstride + half * 16;
      BF16x16 bv;
      bv.u[0] = ((const uint4*)vr)[0];
      bv.u[1] = ((const uint4*)vr)[1];
      acc[s2] = wmma_bf16(ap.v, bv.v, acc[s2]);
    }
    __syncthreads();
#if HAVE_TDM
    cur ^= 1;
#endif
  }
  int b = bh >> 4, h = bh & 15;
#pragma unroll
  for (int r = 0; r < 8; ++r) {
    float inv = 1.f / lrun[r];
    long row = (long)b * SS + q0 + r + half * 8;
#pragma unroll
    for (int s2 = 0; s2 < 4; ++s2)
      ctxb[row * DD + h * HD + s2 * 16 + nlo] = (bf16)(acc[s2][r] * inv);
  }
}

// ---------------------------------------------------------------------------
// Router: softmax over E=8, top-2, normalize, bucket tokens per expert.
// ---------------------------------------------------------------------------
__global__ void __launch_bounds__(128) router_kernel(
    const float* __restrict__ h2f, const float* __restrict__ rw, const float* __restrict__ rb,
    float* __restrict__ combine, int* __restrict__ eidx, int* __restrict__ counts,
    int* __restrict__ toklist) {
  int lane = threadIdx.x & 31;
  int wave = threadIdx.x >> 5;
  int t = blockIdx.x * 4 + wave;
  if (t >= TT) return;
  const float* hr = h2f + (long)t * DD;
  float lg[EE];
#pragma unroll
  for (int e = 0; e < EE; ++e) lg[e] = 0.f;
  for (int d = lane; d < DD; d += 32) {
    float xv = hr[d];
#pragma unroll
    for (int e = 0; e < EE; ++e) lg[e] += xv * rw[d * EE + e];
  }
#pragma unroll
  for (int e = 0; e < EE; ++e)
    for (int off = 16; off > 0; off >>= 1) lg[e] += __shfl_xor(lg[e], off);
  if (lane == 0) {
    float mx = -1e30f;
#pragma unroll
    for (int e = 0; e < EE; ++e) { lg[e] += rb[e]; mx = fmaxf(mx, lg[e]); }
    float p[EE]; float sum = 0.f;
#pragma unroll
    for (int e = 0; e < EE; ++e) { p[e] = __expf(lg[e] - mx); sum += p[e]; }
    float inv = 1.f / sum;
#pragma unroll
    for (int e = 0; e < EE; ++e) p[e] *= inv;
    int i0 = 0;
#pragma unroll
    for (int e = 1; e < EE; ++e) if (p[e] > p[i0]) i0 = e;
    int i1 = (i0 == 0) ? 1 : 0;
#pragma unroll
    for (int e = 0; e < EE; ++e) if (e != i0 && p[e] > p[i1]) i1 = e;
    float wn = p[i0] + p[i1];
    float w0 = p[i0] / wn, w1v = p[i1] / wn;
#pragma unroll
    for (int e = 0; e < EE; ++e)
      combine[t * EE + e] = (e == i0) ? w0 : ((e == i1) ? w1v : 0.f);
    eidx[t * 2] = i0; eidx[t * 2 + 1] = i1;
    int s0 = atomicAdd(&counts[i0], 1); toklist[i0 * TT + s0] = t;
    int s1 = atomicAdd(&counts[i1], 1); toklist[i1 * TT + s1] = t;
  }
}

// ---------------------------------------------------------------------------
// Fused routed MoE.
// ---------------------------------------------------------------------------
__global__ void __launch_bounds__(512) moe_kernel(
    const bf16* __restrict__ h2b, const bf16* __restrict__ w1t, const bf16* __restrict__ w2t,
    const float* __restrict__ b1, const float* __restrict__ b2,
    const float* __restrict__ combine, const int* __restrict__ eidx,
    const int* __restrict__ counts, const int* __restrict__ toklist,
    float* __restrict__ moe2) {
  const int e = blockIdx.y;
  const int mt = blockIdx.x;
  const int cnt = counts[e];
  if (mt * 16 >= cnt) return;
  const int lane = threadIdx.x & 31;
  const int wave = threadIdx.x >> 5;
  const int half = lane >> 4;
  const int nlo  = lane & 15;
  const int kset = half * 8;
  __shared__ bf16 hid[16 * 256];
  __shared__ int toks[16];
  if (threadIdx.x < 16) {
    int sl = mt * 16 + threadIdx.x;
    toks[threadIdx.x] = (sl < cnt) ? toklist[e * TT + sl] : -1;
  }
  __syncthreads();
  int myrow = toks[nlo];
  const bf16* Arow = h2b + (long)(myrow < 0 ? 0 : myrow) * DD;
  const bf16* w1e = w1t + (long)e * FF * DD;
  const bf16* w2e = w2t + (long)e * DD * FF;
  const v8f vz = {0.f,0.f,0.f,0.f,0.f,0.f,0.f,0.f};
  v8f oacc[4] = {vz, vz, vz, vz};
  for (int fb = 0; fb < FF; fb += 256) {
    int f0 = fb + wave * 16;
    v8f hacc = vz;
    const bf16* brow1 = w1e + (long)(f0 + nlo) * DD + half * 16;
    for (int k0 = 0; k0 < DD; k0 += 32) {
      BF16x16 au, bu;
      au.u[0] = *(const uint4*)(Arow + k0 + 2 * kset);
      au.u[1] = *(const uint4*)(Arow + k0 + 8 + 2 * kset);
      bu.u[0] = ((const uint4*)(brow1 + k0))[0];
      bu.u[1] = ((const uint4*)(brow1 + k0))[1];
      hacc = wmma_bf16(au.v, bu.v, hacc);
    }
#pragma unroll
    for (int r = 0; r < 8; ++r) {
      int rowl = r + half * 8;
      float hv = hacc[r] + b1[e * FF + f0 + nlo];
      hv = gelu_tanh(hv);
      hid[rowl * 256 + wave * 16 + nlo] = (bf16)hv;
    }
    __syncthreads();
    const bf16* prow = hid + nlo * 256;
    for (int kk = 0; kk < 256; kk += 32) {
      BF16x16 ap;
      ap.u[0] = *(const uint4*)(prow + kk + 2 * kset);
      ap.u[1] = *(const uint4*)(prow + kk + 8 + 2 * kset);
#pragma unroll
      for (int s2 = 0; s2 < 4; ++s2) {
        int d = wave * 64 + s2 * 16 + nlo;
        const bf16* brow2 = w2e + (long)d * FF + fb + kk + half * 16;
        BF16x16 bu;
        bu.u[0] = ((const uint4*)brow2)[0];
        bu.u[1] = ((const uint4*)brow2)[1];
        oacc[s2] = wmma_bf16(ap.v, bu.v, oacc[s2]);
      }
    }
    __syncthreads();
  }
#pragma unroll
  for (int r = 0; r < 8; ++r) {
    int rowl = r + half * 8;
    int tok = toks[rowl];
    if (tok < 0) continue;
    float cw = combine[tok * EE + e];
    int rank = (eidx[tok * 2] == e) ? 0 : 1;
#pragma unroll
    for (int s2 = 0; s2 < 4; ++s2) {
      int d = wave * 64 + s2 * 16 + nlo;
      moe2[((long)rank * TT + tok) * DD + d] = cw * (oacc[s2][r] + b2[e * DD + d]);
    }
  }
}

__global__ void finalize_kernel(const float* __restrict__ x1, const float* __restrict__ moe2,
                                const int* __restrict__ counts, float* __restrict__ out) {
  long n = (long)TT * DD;
  long stride = (long)gridDim.x * blockDim.x;
  for (long i = (long)blockIdx.x * blockDim.x + threadIdx.x; i < n; i += stride)
    out[i] = x1[i] + moe2[i] + moe2[n + i];
  if (blockIdx.x == 0 && threadIdx.x < EE)
    out[n + threadIdx.x] = (float)counts[threadIdx.x];
}

// ---------------------------------------------------------------------------
// Host launcher
// ---------------------------------------------------------------------------
extern "C" void kernel_launch(void* const* d_in, const int* in_sizes, int n_in,
                              void* d_out, int out_size, void* d_ws, size_t ws_size,
                              hipStream_t stream) {
  const float* x     = (const float*)d_in[0];
  const float* ln1_g = (const float*)d_in[2];
  const float* ln1_b = (const float*)d_in[3];
  const float* w_qkv = (const float*)d_in[4];
  const float* b_qkv = (const float*)d_in[5];
  const float* w_o   = (const float*)d_in[6];
  const float* b_o   = (const float*)d_in[7];
  const float* ln2_g = (const float*)d_in[8];
  const float* ln2_b = (const float*)d_in[9];
  const float* r_w   = (const float*)d_in[10];
  const float* r_b   = (const float*)d_in[11];
  const float* w1    = (const float*)d_in[12];
  const float* b1    = (const float*)d_in[13];
  const float* w2    = (const float*)d_in[14];
  const float* b2    = (const float*)d_in[15];
  float* out = (float*)d_out;

  char* p = (char*)d_ws;
  auto take = [&](size_t bytes) -> void* {
    void* r = (void*)p;
    p += (bytes + 255) & ~(size_t)255;
    return r;
  };
  bf16*  wqkv_b = (bf16*)take((size_t)3 * DD * DD * 2);
  bf16*  wo_b   = (bf16*)take((size_t)DD * DD * 2);
  bf16*  w1t_b  = (bf16*)take((size_t)EE * DD * FF * 2);
  bf16*  w2t_b  = (bf16*)take((size_t)EE * FF * DD * 2);
  bf16*  h1_b   = (bf16*)take((size_t)TT * DD * 2);
  float* qkv_f  = (float*)take((size_t)TT * 3 * DD * 4);
  bf16*  qh     = (bf16*)take((size_t)BB * HH * SS * HD * 2);
  bf16*  kh     = (bf16*)take((size_t)BB * HH * SS * HD * 2);
  bf16*  vT     = (bf16*)take((size_t)BB * HH * HD * SS * 2);
  bf16*  ctx_b  = (bf16*)take((size_t)TT * DD * 2);
  float* x1     = (float*)take((size_t)TT * DD * 4);
  float* h2_f   = (float*)take((size_t)TT * DD * 4);
  bf16*  h2_b   = (bf16*)take((size_t)TT * DD * 2);
  float* comb   = (float*)take((size_t)TT * EE * 4);
  int*   eidx   = (int*)take((size_t)TT * 2 * 4);
  int*   counts = (int*)take((size_t)EE * 4);
  int*   tokl   = (int*)take((size_t)EE * TT * 4);
  float* moe2   = (float*)take((size_t)2 * TT * DD * 4);

  init_counts_kernel<<<1, 32, 0, stream>>>(counts);

  {
    long n = (long)3 * DD * DD;
    cvt_bf16_kernel<<<(int)((n + 255) / 256), 256, 0, stream>>>(w_qkv, wqkv_b, n);
    n = (long)DD * DD;
    cvt_bf16_kernel<<<(int)((n + 255) / 256), 256, 0, stream>>>(w_o, wo_b, n);
    dim3 g1(8192, EE);
    cvt_t_bf16_kernel<<<g1, 256, 0, stream>>>(w1, w1t_b, DD, FF);
    cvt_t_bf16_kernel<<<g1, 256, 0, stream>>>(w2, w2t_b, FF, DD);
  }

  ln_kernel<<<TT, 128, 0, stream>>>(x, ln1_g, ln1_b, h1_b, nullptr);

  {
    dim3 g(TT / 16, (3 * DD) / 256);
    gemm_abt_kernel<<<g, 128, 0, stream>>>(h1_b, wqkv_b, b_qkv, nullptr, qkv_f, DD, 3 * DD);
  }

  split_qkv_kernel<<<16384, 256, 0, stream>>>(qkv_f, qh, kh, vT);

  {
    dim3 g(BB * HH, SS / 16);
    attn_kernel<<<g, 32, 0, stream>>>(qh, kh, vT, ctx_b);
  }

  {
    dim3 g(TT / 16, DD / 256);
    gemm_abt_kernel<<<g, 128, 0, stream>>>(ctx_b, wo_b, b_o, x, x1, DD, DD);
  }

  ln_kernel<<<TT, 128, 0, stream>>>(x1, ln2_g, ln2_b, h2_b, h2_f);

  router_kernel<<<TT / 4, 128, 0, stream>>>(h2_f, r_w, r_b, comb, eidx, counts, tokl);

  {
    dim3 g(TT / 16, EE);
    moe_kernel<<<g, 512, 0, stream>>>(h2_b, w1t_b, w2t_b, b1, b2, comb, eidx, counts, tokl, moe2);
  }

  finalize_kernel<<<16384, 256, 0, stream>>>(x1, moe2, counts, out);
}
